// QSA_48266842472938
// MI455X (gfx1250) — matrix operand order repaired and checked by reading, not compiled, not run on hardware
//
#include <hip/hip_runtime.h>
#include <hip/hip_bf16.h>

#define NW 6

typedef __attribute__((ext_vector_type(16))) _Float16 v16h;
typedef __attribute__((ext_vector_type(8)))  _Float16 v8h;
typedef __attribute__((ext_vector_type(8)))  float    v8f;
typedef __attribute__((ext_vector_type(4)))  int      v4i;
typedef __attribute__((ext_vector_type(8)))  int      v8i;

// One-instruction butterfly exchange: ds_swizzle_b32, group-of-32 xor mode.
// offset encoding: and_mask=0x1F (bits 4:0), or_mask=0 (9:5), xor_mask (14:10).
#define XSWZ(v, imm) __int_as_float(__builtin_amdgcn_ds_swizzle(__float_as_int(v), (imm)))

// ---------------------------------------------------------------------------
// Kernel 1: build the 8 quadratic-form matrices A_m (64x64 each) from the
// circuit parameters, stored as a column-major f16 B matrix: B[n*64 + k],
// n = m*64 + j, value = A_m[k][j].  m: 0=q(w0), 1=k(w0), 2..7=v(w0..5).
// 3 blocks (q,k,v) x 64 threads; thread c simulates circuit on basis state c.
// ---------------------------------------------------------------------------
__global__ void qsa_build_B(const float* __restrict__ pq,
                            const float* __restrict__ pk,
                            const float* __restrict__ pv,
                            _Float16* __restrict__ bws) {
  __shared__ float sre[64 * 65];
  __shared__ float sim[64 * 65];
  const int p = blockIdx.x;          // 0=q, 1=k, 2=v
  const int c = threadIdx.x;         // column 0..63
  const float* prm = (p == 0) ? pq : (p == 1) ? pk : pv;
  float* re = &sre[c * 65];
  float* im = &sim[c * 65];
  for (int k = 0; k < 64; ++k) { re[k] = (k == c) ? 1.f : 0.f; im[k] = 0.f; }

  // wire j <-> bit (5-j)  (wire 0 is most-significant in flattened index)
  for (int j = 0; j < NW; ++j) {
    const int bit = 1 << (5 - j);
    {   // RX(rx0[j])
      float th = prm[0 * NW + j] * 0.5f, cc = __cosf(th), ss = __sinf(th);
      for (int k = 0; k < 64; ++k) if (!(k & bit)) {
        int k1 = k | bit;
        float r0 = re[k], i0 = im[k], r1 = re[k1], i1 = im[k1];
        re[k]  =  cc * r0 + ss * i1;  im[k]  =  cc * i0 - ss * r1;
        re[k1] =  ss * i0 + cc * r1;  im[k1] = -ss * r0 + cc * i1;
      }
    }
    {   // RY(ry0[j])
      float th = prm[1 * NW + j] * 0.5f, cc = __cosf(th), ss = __sinf(th);
      for (int k = 0; k < 64; ++k) if (!(k & bit)) {
        int k1 = k | bit;
        float r0 = re[k], i0 = im[k], r1 = re[k1], i1 = im[k1];
        re[k]  = cc * r0 - ss * r1;  im[k]  = cc * i0 - ss * i1;
        re[k1] = ss * r0 + cc * r1;  im[k1] = ss * i0 + cc * i1;
      }
    }
  }
  for (int j = 0; j < NW; ++j) {     // CNOT ring: control j, target (j+1)%6
    const int cb = 1 << (5 - j), tb = 1 << (5 - ((j + 1) % NW));
    for (int k = 0; k < 64; ++k) if ((k & cb) && !(k & tb)) {
      int k1 = k | tb;
      float tr = re[k], ti = im[k];
      re[k] = re[k1]; im[k] = im[k1];
      re[k1] = tr;    im[k1] = ti;
    }
  }
  for (int j = 0; j < NW; ++j) {     // RY(ry1[j])
    const int bit = 1 << (5 - j);
    float th = prm[2 * NW + j] * 0.5f, cc = __cosf(th), ss = __sinf(th);
    for (int k = 0; k < 64; ++k) if (!(k & bit)) {
      int k1 = k | bit;
      float r0 = re[k], i0 = im[k], r1 = re[k1], i1 = im[k1];
      re[k]  = cc * r0 - ss * r1;  im[k]  = cc * i0 - ss * i1;
      re[k1] = ss * r0 + cc * r1;  im[k1] = ss * i0 + cc * i1;
    }
  }
  __syncthreads();

  // A_w[i][j] = sum_k d(k,w) * (ReU[k][i]ReU[k][j] + ImU[k][i]ImU[k][j])
  const int i = c;   // this thread produces row i (K index of B)
  for (int j = 0; j < 64; ++j) {
    float acc[6] = {0, 0, 0, 0, 0, 0};
    for (int k = 0; k < 64; ++k) {
      float pr = sre[i * 65 + k] * sre[j * 65 + k] +
                 sim[i * 65 + k] * sim[j * 65 + k];
#pragma unroll
      for (int w = 0; w < 6; ++w) acc[w] += ((k >> (5 - w)) & 1) ? -pr : pr;
    }
    if (p < 2) {
      int n = p * 64 + j;
      bws[n * 64 + i] = (_Float16)acc[0];
    } else {
      for (int w = 0; w < 6; ++w) {
        int n = (2 + w) * 64 + j;
        bws[n * 64 + i] = (_Float16)acc[w];
      }
    }
  }
}

// ---------------------------------------------------------------------------
// Kernel 2: TDM-staged WMMA GEMM + in-register contraction.
// The whole 64x512 f16 B panel is DMA'd into LDS by a single
// tensor_load_to_lds (TDM) with pad_enable inserting 16B every 128B so each
// 64-half column has a 144B stride (bank-conflict-free ds reads).
// One wave owns 16 rows of X; two v_wmma_f32_16x16x32_f16 per 16x16 D tile;
// contraction against X in-register; ds_swizzle butterfly reduction over the
// 16-lane N group.  Z layout: Z[row*8 + m].
// ---------------------------------------------------------------------------
__global__ void __launch_bounds__(256)
qsa_gemm_z(const float* __restrict__ X, const _Float16* __restrict__ B,
           float* __restrict__ Z) {
  // 512 columns x 72 halves (64 data + 8 pad) = 73728 B (only LDS object -> offset 0)
  __shared__ _Float16 sB[512 * 72];

  const int tid    = threadIdx.x;
  const int lane   = tid & 31;
  const int waveid = tid >> 5;

  if (waveid == 0) {
    // ---- Tensor DMA descriptor (D#): 1D tile of 32768 f16 elements ----
    unsigned long long ga = (unsigned long long)B;
    v4i g0;
    g0.x = 1;                                   // count=1, no gather
    g0.y = 0;                                   // lds_addr = 0 (sB)
    g0.z = (int)(unsigned)ga;                   // global_addr[31:0]
    g0.w = (int)((((unsigned)(ga >> 32)) & 0x01FFFFFFu) | 0x80000000u); // addr[56:32] | type=2
    v8i g1;
    g1[0] = (1 << 16)        // data_size = 2 bytes
          | (1 << 20)        // pad_enable
          | (4 << 22)        // pad_interval: 32 DWORDs (=128B) before pad
          | (3 << 25);       // pad_amount: 4 DWORDs (=16B)
    g1[1] = (int)0x80000000u;    // tensor_dim0 = 32768 (bits 79:48)
    g1[2] = 0x00010000;          // tensor_dim1 = 1     (bits 111:80)
    g1[3] = (int)0x80000000u;    // tile_dim0   = 32768 (bits 127:112)
    g1[4] = 1;                   // tile_dim1 = 1, tile_dim2 = 0
    g1[5] = 32768;               // tensor_dim0_stride[31:0]
    g1[6] = 0;                   // stride high bits / dim1_stride low
    g1[7] = 0;
    // NOTE: pass &sB as an (unused) operand so the LDS object's address
    // escapes into the asm -- otherwise alias analysis proves sB is never
    // stored and folds all LDS reads to undef (observed in round 2).
    asm volatile("tensor_load_to_lds %0, %1"
                 :: "s"(g0), "s"(g1), "v"((void*)sB)
                 : "memory");
    __builtin_amdgcn_s_wait_tensorcnt(0);
  }
  __syncthreads();

  const int tile    = blockIdx.x * 8 + waveid;
  const int rowbase = tile * 16;
  const int h = lane >> 4;       // K-half select
  const int M = lane & 15;       // A-matrix row within tile
  const int N = lane & 15;       // B/D column within tile

  // --- A fragments (ISA 16-bit A 16x32 layout: pairs, +8h, p>=4 -> +16) ---
  v16h a0, a1;
  const float* xr = X + (size_t)(rowbase + M) * 64;
#pragma unroll
  for (int p = 0; p < 8; ++p) {
    int kl = (p < 4 ? 2 * p : 16 + 2 * (p - 4)) + 8 * h;
    float2 v0 = *(const float2*)(xr + kl);
    float2 v1 = *(const float2*)(xr + 32 + kl);
    a0[2 * p] = (_Float16)v0.x;  a0[2 * p + 1] = (_Float16)v0.y;
    a1[2 * p] = (_Float16)v1.x;  a1[2 * p + 1] = (_Float16)v1.y;
  }

  float nrm[8] = {1, 1, 1, 1, 1, 1, 1, 1};
#pragma unroll
  for (int m = 0; m < 8; ++m) {
    float zacc[8] = {0, 0, 0, 0, 0, 0, 0, 0};
    float nacc[8] = {0, 0, 0, 0, 0, 0, 0, 0};
#pragma unroll
    for (int jt = 0; jt < 4; ++jt) {
      const int n = m * 64 + jt * 16 + N;
      const _Float16* col = sB + n * 72;      // padded column (144B stride)
      // fragment s: halves K = 32s + 8h + [0..8) then 32s + 16 + 8h + [0..8) (16B aligned)
      v8h b0l = *(const v8h*)(col + 8 * h);
      v8h b0h = *(const v8h*)(col + 16 + 8 * h);
      v8h b1l = *(const v8h*)(col + 32 + 8 * h);
      v8h b1h = *(const v8h*)(col + 32 + 16 + 8 * h);
      v16h b0 = __builtin_shufflevector(b0l, b0h, 0,1,2,3,4,5,6,7,8,9,10,11,12,13,14,15);
      v16h b1 = __builtin_shufflevector(b1l, b1h, 0,1,2,3,4,5,6,7,8,9,10,11,12,13,14,15);
      v8f c = {};
      c = __builtin_amdgcn_wmma_f32_16x16x32_f16(false, a0, false, b0,
                                                 (short)0, c, false, false);
      c = __builtin_amdgcn_wmma_f32_16x16x32_f16(false, a1, false, b1,
                                                 (short)0, c, false, false);
#pragma unroll
      for (int r = 0; r < 8; ++r) {  // D element r: row M=r+8h, col N
        float xv = X[(size_t)(rowbase + r + 8 * h) * 64 + jt * 16 + N];
        zacc[r] += xv * c[r];
        if (m == 0) nacc[r] += xv * xv;
      }
    }
    if (m == 0) {
#pragma unroll
      for (int r = 0; r < 8; ++r) {
        float v = nacc[r];
        v += XSWZ(v, 0x041F);
        v += XSWZ(v, 0x081F);
        v += XSWZ(v, 0x101F);
        v += XSWZ(v, 0x201F);
        nrm[r] = v;
      }
    }
#pragma unroll
    for (int r = 0; r < 8; ++r) {
      float v = zacc[r];
      v += XSWZ(v, 0x041F);
      v += XSWZ(v, 0x081F);
      v += XSWZ(v, 0x101F);
      v += XSWZ(v, 0x201F);
      zacc[r] = v;
    }
    if (N == 0) {
#pragma unroll
      for (int r = 0; r < 8; ++r)
        Z[(size_t)(rowbase + r + 8 * h) * 8 + m] = zacc[r] / nrm[r];
    }
  }
}

// ---------------------------------------------------------------------------
// Kernel 3a: per-batch max/min of zk (for stable softmax of rank-1 scores).
// ---------------------------------------------------------------------------
__global__ void qsa_minmax(const float* __restrict__ Z, float* __restrict__ mm) {
  __shared__ float smx[256], smn[256];
  const int b = blockIdx.x, tid = threadIdx.x;
  float mx = -1e30f, mn = 1e30f;
  for (int s = tid; s < 2048; s += 256) {
    float v = Z[(size_t)(b * 2048 + s) * 8 + 1];
    mx = fmaxf(mx, v); mn = fminf(mn, v);
  }
  smx[tid] = mx; smn[tid] = mn; __syncthreads();
  for (int off = 128; off > 0; off >>= 1) {
    if (tid < off) {
      smx[tid] = fmaxf(smx[tid], smx[tid + off]);
      smn[tid] = fminf(smn[tid], smn[tid + off]);
    }
    __syncthreads();
  }
  if (tid == 0) { mm[b * 2] = smx[0]; mm[b * 2 + 1] = smn[0]; }
}

// ---------------------------------------------------------------------------
// Kernel 3b: rank-1 attention. scores[b,t,s] = sqrt(6)*zq[t]*zk[s].
// One wave per (b,t): streaming softmax-weighted sum of v over s;
// full-wave ds_swizzle butterfly reduction.
// ---------------------------------------------------------------------------
__global__ void __launch_bounds__(256)
qsa_attn(const float* __restrict__ Z, const float* __restrict__ mm,
         float* __restrict__ out) {
  const int tid = threadIdx.x, lane = tid & 31, wv = tid >> 5;
  const int b = blockIdx.y;
  const int t = blockIdx.x * 8 + wv;
  const size_t zb = (size_t)b * 2048;
  const float a = 2.449489742783178f * Z[(zb + t) * 8 + 0];  // sqrt(6)*zq
  const float mval = (a >= 0.f) ? a * mm[b * 2] : a * mm[b * 2 + 1];
  float wsum = 0.f, acc[6] = {0, 0, 0, 0, 0, 0};
  __builtin_prefetch(&Z[(zb + lane) * 8], 0, 0);
  for (int s = lane; s < 2048; s += 32) {
    const float* zp = &Z[(zb + s) * 8];
    float w = __expf(a * zp[1] - mval);
    wsum += w;
#pragma unroll
    for (int d = 0; d < 6; ++d) acc[d] += w * zp[2 + d];
  }
  wsum += XSWZ(wsum, 0x041F);
  wsum += XSWZ(wsum, 0x081F);
  wsum += XSWZ(wsum, 0x101F);
  wsum += XSWZ(wsum, 0x201F);
  wsum += XSWZ(wsum, 0x401F);
#pragma unroll
  for (int d = 0; d < 6; ++d) {
    float v = acc[d];
    v += XSWZ(v, 0x041F);
    v += XSWZ(v, 0x081F);
    v += XSWZ(v, 0x101F);
    v += XSWZ(v, 0x201F);
    v += XSWZ(v, 0x401F);
    acc[d] = v;
  }
  if (lane == 0) {
    float inv = 1.f / wsum;
#pragma unroll
    for (int d = 0; d < 6; ++d)
      out[((size_t)b * 2048 + t) * 6 + d] = acc[d] * inv;
  }
}

// ---------------------------------------------------------------------------
extern "C" void kernel_launch(void* const* d_in, const int* in_sizes, int n_in,
                              void* d_out, int out_size, void* d_ws, size_t ws_size,
                              hipStream_t stream) {
  const float* x  = (const float*)d_in[0];   // (16,2048,64)
  const float* pq = (const float*)d_in[1];   // (3,6)
  const float* pk = (const float*)d_in[2];
  const float* pv = (const float*)d_in[3];

  char* ws = (char*)d_ws;
  _Float16* B = (_Float16*)ws;                                   // 64KB
  float* Z    = (float*)(ws + 65536);                            // 1MB
  float* MM   = (float*)(ws + 65536 + 32768 * 8 * sizeof(float));// 128B
  float* out  = (float*)d_out;

  qsa_build_B<<<dim3(3),        dim3(64),  0, stream>>>(pq, pk, pv, B);
  qsa_gemm_z <<<dim3(256),      dim3(256), 0, stream>>>(x, B, Z);
  qsa_minmax <<<dim3(16),       dim3(256), 0, stream>>>(Z, MM);
  qsa_attn   <<<dim3(256, 16),  dim3(256), 0, stream>>>(Z, MM, out);
}
